// QuantBasicTransformerBlock_44555990729123
// MI455X (gfx1250) — matrix operand order repaired and checked by reading, not compiled, not run on hardware
//
#include <hip/hip_runtime.h>
#include <hip/hip_bf16.h>

// ---------------------------------------------------------------------------
// CDNA5 / gfx1250 quantized transformer block.
// wave32, WMMA f32_16x16x32_f16 for all matmuls (projections, QK^T, P*V),
// GLOBAL_LOAD_ASYNC_TO_LDS_B128 for row-major tile staging.
// ---------------------------------------------------------------------------

typedef _Float16 v16h __attribute__((ext_vector_type(16)));
typedef _Float16 v8h  __attribute__((ext_vector_type(8)));
typedef float    v8f  __attribute__((ext_vector_type(8)));

union F16Frag { v16h v; v8h h[2]; _Float16 e[16]; };
union F16x8   { v8h v; _Float16 e[8]; };

__device__ __forceinline__ v8f zf8() {
    v8f z = {0.f,0.f,0.f,0.f,0.f,0.f,0.f,0.f}; return z;
}
__device__ __forceinline__ v8h zh8() {
    F16x8 t;
#pragma unroll
    for (int i = 0; i < 8; ++i) t.e[i] = (_Float16)0.f;
    return t.v;
}

__device__ __forceinline__ v8f wmma_f16(const F16Frag& a, const F16Frag& b, v8f c) {
    return __builtin_amdgcn_wmma_f32_16x16x32_f16(false, a.v, false, b.v,
                                                  (short)0, c, false, false);
}

// async 16B-per-lane global -> LDS copy (ASYNCcnt-tracked DMA path)
__device__ __forceinline__ void async_g2lds_b128(const _Float16* gsrc, _Float16* lds) {
    unsigned int  loff = (unsigned int)(unsigned long long)(uintptr_t)lds;
    unsigned long long ga = (unsigned long long)(uintptr_t)gsrc;
    asm volatile("global_load_async_to_lds_b128 %0, %1, off"
                 :: "v"(loff), "v"(ga) : "memory");
}
__device__ __forceinline__ void wait_async0() {
    asm volatile("s_wait_asynccnt 0x0" ::: "memory");
}

// fake quant-dequant (UniformAffineQuantizer, n_bits=8)
__device__ __forceinline__ float fq8(float x, float delta, float zp) {
    float xi = rintf(x / delta) + zp;
    xi = fminf(fmaxf(xi, 0.f), 255.f);
    return (xi - zp) * delta;
}

// ---------------------------------------------------------------------------
// fp32 -> f16 convert
// ---------------------------------------------------------------------------
__global__ void cvt_f32_f16_kernel(const float* __restrict__ src,
                                   _Float16* __restrict__ dst, size_t n) {
    size_t i = (size_t)blockIdx.x * 256u + threadIdx.x;
    if (i < n) dst[i] = (_Float16)src[i];
}

// ---------------------------------------------------------------------------
// LayerNorm over D=1024: one block (256 threads) per row; f16 output
// ---------------------------------------------------------------------------
__global__ __launch_bounds__(256)
void ln_kernel(const float* __restrict__ x, const float* __restrict__ g,
               const float* __restrict__ bta, _Float16* __restrict__ out, int Dm) {
    int row = blockIdx.x;
    const float* xr = x + (size_t)row * Dm;
    float s = 0.f, s2 = 0.f;
    for (int i = threadIdx.x; i < Dm; i += 256) {
        float v = xr[i]; s += v; s2 += v * v;
    }
#pragma unroll
    for (int off = 16; off; off >>= 1) {
        s  += __shfl_xor(s,  off);
        s2 += __shfl_xor(s2, off);
    }
    __shared__ float red0[8], red1[8];
    int wv = threadIdx.x >> 5;
    if ((threadIdx.x & 31) == 0) { red0[wv] = s; red1[wv] = s2; }
    __syncthreads();
    s = 0.f; s2 = 0.f;
#pragma unroll
    for (int w = 0; w < 8; ++w) { s += red0[w]; s2 += red1[w]; }
    float inv = 1.f / (float)Dm;
    float mu = s * inv;
    float var = s2 * inv - mu * mu;
    float rstd = rsqrtf(var + 1e-5f);
    for (int i = threadIdx.x; i < Dm; i += 256) {
        out[(size_t)row * Dm + i] = (_Float16)((xr[i] - mu) * rstd * g[i] + bta[i]);
    }
}

// ---------------------------------------------------------------------------
// GEGLU: out[r,c] = a * gelu_exact(g), a = hg[r,c], g = hg[r, Dh + c]
// ---------------------------------------------------------------------------
__global__ void geglu_kernel(const _Float16* __restrict__ hg,
                             _Float16* __restrict__ out, size_t rows, int Dh) {
    size_t i = (size_t)blockIdx.x * 256u + threadIdx.x;
    if (i >= rows * (size_t)Dh) return;
    size_t r = i / (size_t)Dh;
    int c = (int)(i - r * (size_t)Dh);
    float a = (float)hg[r * (size_t)(2 * Dh) + c];
    float gg = (float)hg[r * (size_t)(2 * Dh) + Dh + c];
    float ge = 0.5f * gg * (1.f + erff(gg * 0.70710678118654752f));
    out[i] = (_Float16)(a * ge);
}

// ---------------------------------------------------------------------------
// WMMA GEMM: C(MxN) = A(MxK,f16) * W(KxN,f16) [+bias][+resid]
//   -> optional fp32 out, optional f16 out with optional 8-bit fake-quant.
// Block: 256 threads (8 waves), 128x64 tile, K-step 64.
// A tile staged via GLOBAL_LOAD_ASYNC_TO_LDS_B128 (clamp-addressed rows);
// W tile staged transposed via VGPRs so B frags are ds_load_b128 pairs.
// B fragments are preloaded per K-substep so WMMAs issue back-to-back and
// the waitcnt pass can overlap LDS latency with the matrix pipe.
// ---------------------------------------------------------------------------
__global__ __launch_bounds__(256)
void gemm_f16_wmma(const _Float16* __restrict__ A, const _Float16* __restrict__ W,
                   const float* __restrict__ bias, const float* __restrict__ resid,
                   float* __restrict__ outF, _Float16* __restrict__ outH,
                   const float* __restrict__ dq, float zp,
                   int M, int N, int K) {
    __shared__ __align__(16) _Float16 As[128][72];   // rows x (64 + 8 pad)
    __shared__ __align__(16) _Float16 BsT[64][72];   // transposed: [n][k]

    const int tid  = threadIdx.x;
    const int wave = tid >> 5;
    const int lane = tid & 31;
    const int hlf  = lane >> 4;
    const int m    = lane & 15;
    const int rowbase = blockIdx.x * 128;
    const int nbase   = blockIdx.y * 64;

    v8f acc[4];
#pragma unroll
    for (int c = 0; c < 4; ++c) acc[c] = zf8();

    for (int kb = 0; kb < K; kb += 64) {
        // A tile 128x64: 1024 slots of 8 f16, 4 per thread, async DMA
#pragma unroll
        for (int it = 0; it < 4; ++it) {
            int s = tid + it * 256;
            int r = s >> 3, cb = s & 7;
            int gr = rowbase + r;
            if (gr > M - 1) gr = M - 1;   // clamp: garbage rows never stored
            async_g2lds_b128(A + (size_t)gr * K + kb + cb * 8, &As[r][cb * 8]);
        }
        // W tile 64x64, stored transposed: 512 slots, 2 per thread
#pragma unroll
        for (int it = 0; it < 2; ++it) {
            int s = tid + it * 256;
            int r = s >> 3, cb = s & 7;
            F16x8 val;
            val.v = *(const v8h*)(W + (size_t)(kb + r) * N + nbase + cb * 8);
#pragma unroll
            for (int i = 0; i < 8; ++i) BsT[cb * 8 + i][r] = val.e[i];
        }
        wait_async0();
        __syncthreads();

        const int arow = wave * 16 + m;
#pragma unroll
        for (int ks = 0; ks < 2; ++ks) {
            F16Frag af;
            af.h[0] = *(const v8h*)(&As[arow][ks * 32 + 16 * hlf]);
            af.h[1] = *(const v8h*)(&As[arow][ks * 32 + 16 * hlf + 8]);
            F16Frag bf[4];
#pragma unroll
            for (int c = 0; c < 4; ++c) {
                int bn = c * 16 + m;
                bf[c].h[0] = *(const v8h*)(&BsT[bn][ks * 32 + 16 * hlf]);
                bf[c].h[1] = *(const v8h*)(&BsT[bn][ks * 32 + 16 * hlf + 8]);
            }
#pragma unroll
            for (int c = 0; c < 4; ++c)
                acc[c] = wmma_f16(af, bf[c], acc[c]);
        }
        __syncthreads();
    }

    // Epilogue: lane holds column c*16+m, rows r+8*half
#pragma unroll
    for (int c = 0; c < 4; ++c) {
        int gn = nbase + c * 16 + m;
        float bv = bias ? bias[gn] : 0.f;
#pragma unroll
        for (int r = 0; r < 8; ++r) {
            int gr = rowbase + wave * 16 + hlf * 8 + r;
            if (gr >= M) continue;
            float v = acc[c][r] + bv;
            size_t idx = (size_t)gr * N + gn;
            if (resid) v += resid[idx];
            if (outF) outF[idx] = v;
            if (outH) {
                float q = v;
                if (dq) q = fq8(q, *dq, zp);
                outH[idx] = (_Float16)q;
            }
        }
    }
}

// ---------------------------------------------------------------------------
// Fused flash-style attention, one (b,h) per blockIdx.x, 64 q rows per
// blockIdx.y. q/k/v are pre-fake-quantized f16; probs fake-quantized (zp=0).
// Q and K tiles staged via async DMA; V staged transposed via VGPRs.
// ---------------------------------------------------------------------------
__global__ __launch_bounds__(128)
void attn_wmma_kernel(const _Float16* __restrict__ Q, const _Float16* __restrict__ Km,
                      const _Float16* __restrict__ Vm, _Float16* __restrict__ O,
                      const float* __restrict__ dwPtr,
                      int Nq, int Nkv, int Dm, int Hh, size_t kvBatchStride) {
    __shared__ __align__(16) _Float16 Qs[64][72];
    __shared__ __align__(16) _Float16 Ks[32][72];
    __shared__ __align__(16) _Float16 Vs[64][40];     // transposed [dh][kv]
    __shared__ __align__(16) _Float16 Ps[4][16][40];  // per-wave prob tile

    const int bh = blockIdx.x;
    const int b = bh / Hh, h = bh % Hh;
    const int qbase = blockIdx.y * 64;
    const int tid = threadIdx.x;
    const int wave = tid >> 5, lane = tid & 31;
    const int hlf = lane >> 4, m = lane & 15;
    const float SCALE = 0.125f;  // DH^-0.5, DH=64
    const float dw = *dwPtr;

    const _Float16* Qb = Q  + (size_t)b * Nq * Dm + (size_t)h * 64;
    const _Float16* Kb = Km + (size_t)b * kvBatchStride + (size_t)h * 64;
    const _Float16* Vb = Vm + (size_t)b * kvBatchStride + (size_t)h * 64;

    // Q tile 64x64 -> LDS: 512 slots, 4 per thread, async DMA
#pragma unroll
    for (int it = 0; it < 4; ++it) {
        int s = tid + it * 128;
        int r = s >> 3, cb = s & 7;
        async_g2lds_b128(Qb + (size_t)(qbase + r) * Dm + cb * 8, &Qs[r][cb * 8]);
    }
    wait_async0();
    __syncthreads();

    F16Frag qf[2];
    const int qrow = wave * 16 + m;
#pragma unroll
    for (int ks = 0; ks < 2; ++ks) {
        qf[ks].h[0] = *(const v8h*)(&Qs[qrow][ks * 32 + 16 * hlf]);
        qf[ks].h[1] = *(const v8h*)(&Qs[qrow][ks * 32 + 16 * hlf + 8]);
    }

    v8f o[4];
#pragma unroll
    for (int c = 0; c < 4; ++c) o[c] = zf8();
    float mrow[8], lrow[8];
#pragma unroll
    for (int r = 0; r < 8; ++r) { mrow[r] = -1e30f; lrow[r] = 0.f; }

    for (int j0 = 0; j0 < Nkv; j0 += 32) {
        // K chunk 32x64 row-major, async, clamp-addressed (scores masked below)
#pragma unroll
        for (int it = 0; it < 2; ++it) {
            int s = tid + it * 128;
            int r = s >> 3, cb = s & 7;
            int j = j0 + r;
            if (j > Nkv - 1) j = Nkv - 1;
            async_g2lds_b128(Kb + (size_t)j * Dm + cb * 8, &Ks[r][cb * 8]);
        }
        // V chunk transposed via VGPRs, zero-filled out of range
#pragma unroll
        for (int it = 0; it < 2; ++it) {
            int s = tid + it * 128;
            int r = s >> 3, cb = s & 7;
            F16x8 val; val.v = zh8();
            if (j0 + r < Nkv) val.v = *(const v8h*)(Vb + (size_t)(j0 + r) * Dm + cb * 8);
#pragma unroll
            for (int i = 0; i < 8; ++i) Vs[cb * 8 + i][r] = val.e[i];
        }
        wait_async0();
        __syncthreads();

        // s = q * k^T : preload all 4 K fragments, then 4 WMMAs back-to-back
        F16Frag kf[2][2];
#pragma unroll
        for (int c = 0; c < 2; ++c) {
            int kn = c * 16 + m;
#pragma unroll
            for (int ks = 0; ks < 2; ++ks) {
                kf[c][ks].h[0] = *(const v8h*)(&Ks[kn][ks * 32 + 16 * hlf]);
                kf[c][ks].h[1] = *(const v8h*)(&Ks[kn][ks * 32 + 16 * hlf + 8]);
            }
        }
        v8f sc[2];
#pragma unroll
        for (int c = 0; c < 2; ++c) {
            sc[c] = zf8();
#pragma unroll
            for (int ks = 0; ks < 2; ++ks)
                sc[c] = wmma_f16(qf[ks], kf[c][ks], sc[c]);
        }
        // scale + mask
#pragma unroll
        for (int c = 0; c < 2; ++c) {
            int j = j0 + c * 16 + m;
            bool oob = (j >= Nkv);
#pragma unroll
            for (int r = 0; r < 8; ++r)
                sc[c][r] = oob ? -1e30f : sc[c][r] * SCALE;
        }
        // online softmax row stats (rows r+8*half; reduce across 16-lane half)
        float mcur[8];
#pragma unroll
        for (int r = 0; r < 8; ++r) mcur[r] = fmaxf(sc[0][r], sc[1][r]);
#pragma unroll
        for (int off = 1; off < 16; off <<= 1)
#pragma unroll
            for (int r = 0; r < 8; ++r)
                mcur[r] = fmaxf(mcur[r], __shfl_xor(mcur[r], off));
        float mnew[8], resc[8];
#pragma unroll
        for (int r = 0; r < 8; ++r) {
            mnew[r] = fmaxf(mrow[r], mcur[r]);
            resc[r] = expf(mrow[r] - mnew[r]);
        }
#pragma unroll
        for (int c = 0; c < 2; ++c)
#pragma unroll
            for (int r = 0; r < 8; ++r)
                sc[c][r] = expf(sc[c][r] - mnew[r]);
        float rsum[8];
#pragma unroll
        for (int r = 0; r < 8; ++r) rsum[r] = sc[0][r] + sc[1][r];
#pragma unroll
        for (int off = 1; off < 16; off <<= 1)
#pragma unroll
            for (int r = 0; r < 8; ++r)
                rsum[r] += __shfl_xor(rsum[r], off);
#pragma unroll
        for (int r = 0; r < 8; ++r) {
            lrow[r] = lrow[r] * resc[r] + rsum[r];
            mrow[r] = mnew[r];
        }
#pragma unroll
        for (int c2 = 0; c2 < 4; ++c2)
#pragma unroll
            for (int r = 0; r < 8; ++r) o[c2][r] *= resc[r];

        // fake-quant probs (zp=0) and reshape C-layout -> A-layout via LDS
#pragma unroll
        for (int c = 0; c < 2; ++c)
#pragma unroll
            for (int r = 0; r < 8; ++r) {
                float p = fminf(fmaxf(rintf(sc[c][r] / dw), 0.f), 255.f) * dw;
                Ps[wave][r + 8 * hlf][c * 16 + m] = (_Float16)p;
            }
        __syncthreads();

        // p x V : preload P and all 4 V fragments, then 4 WMMAs back-to-back
        F16Frag pf;
        pf.h[0] = *(const v8h*)(&Ps[wave][m][16 * hlf]);
        pf.h[1] = *(const v8h*)(&Ps[wave][m][16 * hlf + 8]);
        F16Frag vfr[4];
#pragma unroll
        for (int c2 = 0; c2 < 4; ++c2) {
            int vn = c2 * 16 + m;
            vfr[c2].h[0] = *(const v8h*)(&Vs[vn][16 * hlf]);
            vfr[c2].h[1] = *(const v8h*)(&Vs[vn][16 * hlf + 8]);
        }
#pragma unroll
        for (int c2 = 0; c2 < 4; ++c2)
            o[c2] = wmma_f16(pf, vfr[c2], o[c2]);
        __syncthreads();
    }

    // normalize and write [b, row, h*64 + col] as f16
#pragma unroll
    for (int c2 = 0; c2 < 4; ++c2)
#pragma unroll
        for (int r = 0; r < 8; ++r) {
            int row = qbase + wave * 16 + 8 * hlf + r;
            float val = o[c2][r] / lrow[r];
            O[((size_t)b * Nq + row) * Dm + h * 64 + c2 * 16 + m] = (_Float16)val;
        }
}

// ---------------------------------------------------------------------------
// Host-side orchestration
// ---------------------------------------------------------------------------
extern "C" void kernel_launch(void* const* d_in, const int* in_sizes, int n_in,
                              void* d_out, int out_size, void* d_ws, size_t ws_size,
                              hipStream_t stream) {
    (void)in_sizes; (void)n_in; (void)out_size; (void)ws_size;
    const int B = 2, N = 2048, Mctx = 77, D = 1024, H = 16;
    const int BN = B * N;              // 4096
    const int Dff = 8 * D;             // 8192
    const int Dh  = 4 * D;             // 4096

    const float* x      = (const float*)d_in[0];
    const float* ctx    = (const float*)d_in[1];
    const float* ln1_g  = (const float*)d_in[2];
    const float* ln1_b  = (const float*)d_in[3];
    const float* ln2_g  = (const float*)d_in[4];
    const float* ln2_b  = (const float*)d_in[5];
    const float* ln3_g  = (const float*)d_in[6];
    const float* ln3_b  = (const float*)d_in[7];
    const float* a1_wq  = (const float*)d_in[8];
    const float* a1_wk  = (const float*)d_in[9];
    const float* a1_wv  = (const float*)d_in[10];
    const float* a1_wo  = (const float*)d_in[11];
    const float* a1_bo  = (const float*)d_in[12];
    const float* a2_wq  = (const float*)d_in[13];
    const float* a2_wk  = (const float*)d_in[14];
    const float* a2_wv  = (const float*)d_in[15];
    const float* a2_wo  = (const float*)d_in[16];
    const float* a2_bo  = (const float*)d_in[17];
    const float* ff_w1  = (const float*)d_in[18];
    const float* ff_b1  = (const float*)d_in[19];
    const float* ff_w2  = (const float*)d_in[20];
    const float* ff_b2  = (const float*)d_in[21];
    const float* d_q1   = (const float*)d_in[22];
    const float* d_k1   = (const float*)d_in[23];
    const float* d_v1   = (const float*)d_in[24];
    const float* d_w1   = (const float*)d_in[25];
    const float* d_q2   = (const float*)d_in[26];
    const float* d_k2   = (const float*)d_in[27];
    const float* d_v2   = (const float*)d_in[28];
    const float* d_w2   = (const float*)d_in[29];

    // workspace carving (256B aligned)
    char* p = (char*)d_ws;
    auto carve = [&](size_t bytes) -> char* {
        char* r = p; p += (bytes + 255) & ~(size_t)255; return r;
    };
    const size_t DD = (size_t)D * D;
    _Float16* wq1f = (_Float16*)carve(DD * 2);
    _Float16* wk1f = (_Float16*)carve(DD * 2);
    _Float16* wv1f = (_Float16*)carve(DD * 2);
    _Float16* wo1f = (_Float16*)carve(DD * 2);
    _Float16* wq2f = (_Float16*)carve(DD * 2);
    _Float16* wk2f = (_Float16*)carve(DD * 2);
    _Float16* wv2f = (_Float16*)carve(DD * 2);
    _Float16* wo2f = (_Float16*)carve(DD * 2);
    _Float16* w1f  = (_Float16*)carve((size_t)D * Dff * 2);
    _Float16* w2f  = (_Float16*)carve((size_t)Dh * D * 2);
    _Float16* ctxf = (_Float16*)carve((size_t)B * Mctx * D * 2);
    _Float16* hf   = (_Float16*)carve((size_t)BN * D * 2);
    _Float16* qf   = (_Float16*)carve((size_t)BN * D * 2);
    _Float16* kf   = (_Float16*)carve((size_t)BN * D * 2);
    _Float16* vf   = (_Float16*)carve((size_t)BN * D * 2);
    _Float16* aof  = (_Float16*)carve((size_t)BN * D * 2);
    float*    xcur = (float*)   carve((size_t)BN * D * 4);
    _Float16* hgf  = (_Float16*)carve((size_t)BN * Dff * 2);
    _Float16* ggf  = (_Float16*)carve((size_t)BN * Dh * 2);

    auto cvt = [&](const float* s, _Float16* d, size_t n) {
        cvt_f32_f16_kernel<<<dim3((unsigned)((n + 255) / 256)), 256, 0, stream>>>(s, d, n);
    };
    auto gemm = [&](const _Float16* A, const _Float16* W, const float* bias,
                    const float* resid, float* oF, _Float16* oH,
                    const float* dq, float zp, int Mr, int Nc, int Kc) {
        dim3 g((Mr + 127) / 128, Nc / 64);
        gemm_f16_wmma<<<g, 256, 0, stream>>>(A, W, bias, resid, oF, oH, dq, zp,
                                             Mr, Nc, Kc);
    };

    // 0) convert weights + context to f16
    cvt(a1_wq, wq1f, DD); cvt(a1_wk, wk1f, DD); cvt(a1_wv, wv1f, DD); cvt(a1_wo, wo1f, DD);
    cvt(a2_wq, wq2f, DD); cvt(a2_wk, wk2f, DD); cvt(a2_wv, wv2f, DD); cvt(a2_wo, wo2f, DD);
    cvt(ff_w1, w1f, (size_t)D * Dff);
    cvt(ff_w2, w2f, (size_t)Dh * D);
    cvt(ctx, ctxf, (size_t)B * Mctx * D);

    // 1) self-attention: h1 = LN1(x); q/k/v = fq(h1 @ W); flash attn; o-proj + x
    ln_kernel<<<BN, 256, 0, stream>>>(x, ln1_g, ln1_b, hf, D);
    gemm(hf, wq1f, nullptr, nullptr, nullptr, qf, d_q1, 128.f, BN, D, D);
    gemm(hf, wk1f, nullptr, nullptr, nullptr, kf, d_k1, 128.f, BN, D, D);
    gemm(hf, wv1f, nullptr, nullptr, nullptr, vf, d_v1, 128.f, BN, D, D);
    attn_wmma_kernel<<<dim3(B * H, N / 64), 128, 0, stream>>>(
        qf, kf, vf, aof, d_w1, N, N, D, H, (size_t)N * D);
    gemm(aof, wo1f, a1_bo, x, xcur, nullptr, nullptr, 0.f, BN, D, D);

    // 2) cross-attention: h2 = LN2(xcur); q from h2, k/v from context
    ln_kernel<<<BN, 256, 0, stream>>>(xcur, ln2_g, ln2_b, hf, D);
    gemm(hf,   wq2f, nullptr, nullptr, nullptr, qf, d_q2, 128.f, BN, D, D);
    gemm(ctxf, wk2f, nullptr, nullptr, nullptr, kf, d_k2, 128.f, B * Mctx, D, D);
    gemm(ctxf, wv2f, nullptr, nullptr, nullptr, vf, d_v2, 128.f, B * Mctx, D, D);
    attn_wmma_kernel<<<dim3(B * H, N / 64), 128, 0, stream>>>(
        qf, kf, vf, aof, d_w2, N, Mctx, D, H, (size_t)Mctx * D);
    gemm(aof, wo2f, a2_bo, xcur, xcur, nullptr, nullptr, 0.f, BN, D, D);

    // 3) GEGLU FF: h3 = LN3(xcur); hg = h3@W1+b1; gg = a*gelu(g); out = gg@W2+b2+xcur
    ln_kernel<<<BN, 256, 0, stream>>>(xcur, ln3_g, ln3_b, hf, D);
    gemm(hf, w1f, ff_b1, nullptr, nullptr, hgf, nullptr, 0.f, BN, Dff, D);
    {
        size_t tot = (size_t)BN * Dh;
        geglu_kernel<<<dim3((unsigned)((tot + 255) / 256)), 256, 0, stream>>>(
            hgf, ggf, (size_t)BN, Dh);
    }
    gemm(ggf, w2f, ff_b2, xcur, (float*)d_out, nullptr, nullptr, 0.f, BN, D, Dh);
}
